// InterpretableMultiHeadAttention_46858093199746
// MI455X (gfx1250) — matrix operand order repaired and checked by reading, not compiled, not run on hardware
//
#include <hip/hip_runtime.h>
#include <hip/hip_bf16.h>
#include <stdint.h>

// ---------------------------------------------------------------------------
// InterpretableMultiHeadAttention forward for MI455X (gfx1250, wave32, WMMA).
// All matmuls use v_wmma_f32_16x16x32_f16 (f16 inputs, f32 accumulation).
// Memory-bound problem (~100MB @ 23.3TB/s); K tiles staged via the Tensor
// Data Mover (tensor_load_to_lds + s_wait_tensorcnt), scores/softmax/PV in
// a flash-attention loop.
// ---------------------------------------------------------------------------

#define BSZ   4
#define SEQ   2048
#define NH    8
#define DH    64
#define HID   512
#define QKVN  1088          // (2*8+1)*64
#define ROWS  (BSZ*SEQ)     // 8192
#define SCALE 0.125f        // 64^-0.5
#define NEGBIG (-1e30f)

typedef _Float16 h8  __attribute__((ext_vector_type(8)));
typedef _Float16 h16 __attribute__((ext_vector_type(16)));
typedef float    f8  __attribute__((ext_vector_type(8)));
typedef unsigned int v4u __attribute__((ext_vector_type(4)));
typedef int          v4i __attribute__((ext_vector_type(4)));
typedef int          v8i __attribute__((ext_vector_type(8)));

__device__ inline h16 combine_h16(h8 lo, h8 hi) {
  h16 r;
#pragma unroll
  for (int i = 0; i < 8; ++i) { r[i] = lo[i]; r[i + 8] = hi[i]; }
  return r;
}
__device__ inline h16 load_frag_split(const _Float16* p0, const _Float16* p1) {
  return combine_h16(*(const h8*)p0, *(const h8*)p1);
}
__device__ inline h16 load_frag(const _Float16* p) {
  return combine_h16(*(const h8*)p, *(const h8*)(p + 16));
}
__device__ inline f8 wmma_f16(h16 a, h16 b, f8 c) {
  // (neg_a, A, neg_b, B, c_mod, C, reuse_a, reuse_b)
  return __builtin_amdgcn_wmma_f32_16x16x32_f16(false, a, false, b, (short)0, c,
                                                false, false);
}

// ---------------------------------------------------------------------------
// Kernel 1: QKV projection  C(8192x1088) = X(8192x512) @ Wqkv(512x1088)
// f32 inputs staged to LDS as f16; 8 waves/block, block tile 128x64,
// wave tile 32x32 (2x2 WMMA), K loop step 32.  The 64-wide N tile maps to
// exactly one of Q / K / V and one head (region boundaries are 64-aligned),
// so output routing is block-uniform: no per-element branches.
// ---------------------------------------------------------------------------
__global__ __launch_bounds__(256) void qkv_gemm_kernel(
    const float* __restrict__ X, const float* __restrict__ Wqkv,
    _Float16* __restrict__ Qh, _Float16* __restrict__ Kh,
    _Float16* __restrict__ Vh) {
  __shared__ _Float16 Xs[128 * 40];   // row-major, pad 40
  __shared__ _Float16 WsT[64 * 40];   // transposed W tile: [n][k]

  const int tid  = threadIdx.x;
  const int wave = tid >> 5, lane = tid & 31;
  const int g = lane >> 4, ln = lane & 15;
  const int m0b = blockIdx.x * 128;
  const int n0b = blockIdx.y * 64;
  const int wm = wave >> 1, wn = wave & 1;   // waves: 4 in M x 2 in N

  f8 acc[2][2] = {};

  for (int kk = 0; kk < HID; kk += 32) {
    // stage X tile 128x32 (f32 -> f16)
    for (int idx = tid; idx < 128 * 32; idx += 256) {
      int r = idx >> 5, c = idx & 31;
      Xs[r * 40 + c] = (_Float16)X[(size_t)(m0b + r) * HID + kk + c];
    }
    // stage W^T tile 64(n) x 32(k)
    for (int idx = tid; idx < 32 * 64; idx += 256) {
      int r = idx >> 6, c = idx & 63;   // r = k-local, c = n-local
      WsT[c * 40 + r] = (_Float16)Wqkv[(size_t)(kk + r) * QKVN + n0b + c];
    }
    __syncthreads();

    h16 a[2], b[2];
#pragma unroll
    for (int tm = 0; tm < 2; ++tm) {
      const _Float16* base = &Xs[(wm * 32 + tm * 16 + ln) * 40];
      a[tm] = load_frag_split(base + 8 * g, base + 16 + 8 * g);
    }
#pragma unroll
    for (int tn = 0; tn < 2; ++tn) {
      const _Float16* base = &WsT[(wn * 32 + tn * 16 + ln) * 40];
    b[tn] = load_frag(base + 8 * g);
    }
#pragma unroll
    for (int tm = 0; tm < 2; ++tm)
#pragma unroll
      for (int tn = 0; tn < 2; ++tn)
        acc[tm][tn] = wmma_f16(a[tm], b[tn], acc[tm][tn]);
    __syncthreads();
  }

  // block-uniform output routing
  const int region = n0b >> 9;          // 0=Q, 1=K, 2=V
  const int head   = (n0b >> 6) & 7;    // head for Q/K
  _Float16* const dst = (region == 0) ? Qh : (region == 1) ? Kh : Vh;

#pragma unroll
  for (int tm = 0; tm < 2; ++tm)
#pragma unroll
    for (int tn = 0; tn < 2; ++tn)
#pragma unroll
      for (int r = 0; r < 8; ++r) {
        int grow = m0b + wm * 32 + tm * 16 + r + 8 * g;
        int dloc = wn * 32 + tn * 16 + ln;
        int bb = grow >> 11, tt = grow & (SEQ - 1);
        size_t rowsel = (region == 2)
                            ? ((size_t)bb * SEQ + tt)
                            : ((size_t)(bb * NH + head) * SEQ + tt);
        dst[rowsel * DH + dloc] = (_Float16)acc[tm][tn][r];
      }
}

// ---------------------------------------------------------------------------
// Kernel 2: causal flash attention per (b,h).  Block = 8 waves, each wave owns
// 16 queries (block covers 128).  Per 32-key tile:
//   - K tile (32x64 f16) DMA'd into padded LDS by the Tensor Data Mover
//     (tensor_load_to_lds; hardware pads each 32-DWORD row with 8 DWORDs to
//     produce our 80-half stride), completion via s_wait_tensorcnt.
//   - V^T staged manually (TDM cannot transpose).
//   - Q@K^T via WMMA, online softmax (ds_bpermute shfl reductions),
//     P transposed through per-wave LDS, P@V via WMMA.
// ---------------------------------------------------------------------------
__global__ __launch_bounds__(256) void attn_kernel(
    const _Float16* __restrict__ Qh, const _Float16* __restrict__ Kh,
    const _Float16* __restrict__ Vh, float* __restrict__ attn) {
  __shared__ _Float16 Ks[32 * 80];       // [key][d], pad 80 (TDM-padded)
  __shared__ _Float16 Vt[64 * 48];       // [d][key], pad 48
  __shared__ _Float16 Ps[8 * 16 * 40];   // per-wave 16x32 P patch, pad 40

  const int tid  = threadIdx.x;
  const int wave = tid >> 5, lane = tid & 31;
  const int g = lane >> 4, ln = lane & 15;
  const int bh = blockIdx.x;             // b*8 + h
  const int b  = bh >> 3;
  const int qb = blockIdx.y;
  const int q0 = qb * 128 + wave * 16;
  const size_t qkBase = (size_t)bh * SEQ * DH;
  const size_t vBase  = (size_t)b * SEQ * DH;

  // Q A-fragments: two d-chunks of 32
  h16 qa[2];
#pragma unroll
  for (int dt = 0; dt < 2; ++dt) {
    const _Float16* base = Qh + qkBase + (size_t)(q0 + ln) * DH + dt * 32;
    qa[dt] = load_frag_split(base + 8 * g, base + 16 + 8 * g);
  }

  float mI[8], lI[8];
  f8 o[4] = {};
#pragma unroll
  for (int r = 0; r < 8; ++r) { mI[r] = NEGBIG; lI[r] = 0.f; }

  const int ntiles = qb * 4 + 4;         // same for all waves (barrier parity)
  _Float16* myPs = &Ps[wave * 16 * 40];
  const unsigned ksLds = (unsigned)(uintptr_t)(void*)&Ks[0];

  for (int kt = 0; kt < ntiles; ++kt) {
    const int k0 = kt * 32;

    // ---- TDM: DMA K tile (32 rows x 64 halfs) into padded LDS (wave 0) ----
    if (wave == 0) {
      unsigned long long gaddr =
          (unsigned long long)(uintptr_t)(const void*)(Kh + qkBase +
                                                       (size_t)k0 * DH);
      v4u g0;
      g0[0] = 1u;                                   // count=1, gather off
      g0[1] = ksLds;                                // lds_addr
      g0[2] = (unsigned)(gaddr & 0xFFFFFFFFu);      // global_addr[31:0]
      g0[3] = (unsigned)((gaddr >> 32) & 0x1FFFFFFu) | (2u << 30);  // type=2
      v8i g1;
      g1[0] = (1 << 16)        // data_size = 2 bytes
            | (1 << 20)        // pad_enable
            | (4 << 22)        // pad_interval: 32 DWORDs (one 64-half row)
            | (7 << 25);       // pad_amount: 8 DWORDs (16 halfs)
      g1[1] = (int)(64u << 16);    // tensor_dim0 = 64 (low16 @ bits 63:48)
      g1[2] = (int)(2048u << 16);  // tensor_dim1 = 2048 (low16 @ bits 95:80)
      g1[3] = (int)(64u << 16);    // tile_dim0 = 64 (bits 127:112)
      g1[4] = 32;                  // tile_dim1 = 32 (bits 143:128)
      g1[5] = 64;                  // tensor_dim0_stride = 64
      g1[6] = 0;
      g1[7] = 0;
      v4i gz = {0, 0, 0, 0};
#if defined(__clang_major__) && (__clang_major__ >= 23)
      v8i gz8 = {0, 0, 0, 0, 0, 0, 0, 0};
      __builtin_amdgcn_tensor_load_to_lds(g0, g1, gz, gz, gz8, 0);
#else
      __builtin_amdgcn_tensor_load_to_lds(g0, g1, gz, gz, 0);
#endif
      __builtin_amdgcn_s_wait_tensorcnt(0);
    }

    // ---- V^T tile staged manually (TDM cannot transpose) ----
    for (int idx = tid; idx < 2048; idx += 256) {
      int d = idx >> 5, kl = idx & 31;
      Vt[d * 48 + kl] = Vh[vBase + (size_t)(k0 + kl) * DH + d];
    }
    if (kt + 1 < ntiles)  // pull next K tile toward L2 (global_prefetch_b8)
      __builtin_prefetch(Kh + qkBase + (size_t)(k0 + 32 + (tid & 31)) * DH, 0, 1);
    __syncthreads();

    // scores: two 16-key column tiles, each accumulated over two d-chunks
    f8 s[2];
#pragma unroll
    for (int j = 0; j < 2; ++j) {
      f8 c = {};
      h16 b0 = load_frag(&Ks[(j * 16 + ln) * 80 + 8 * g]);        // d 0..31
      h16 b1 = load_frag(&Ks[(j * 16 + ln) * 80 + 32 + 8 * g]);   // d 32..63
      c = wmma_f16(qa[0], b0, c);
      c = wmma_f16(qa[1], b1, c);
      s[j] = c;
    }

    // scale + causal mask + online softmax
    float p0[8], p1[8];
#pragma unroll
    for (int r = 0; r < 8; ++r) {
      int qg  = q0 + r + 8 * g;
      int kg0 = k0 + ln, kg1 = kg0 + 16;
      float v0 = s[0][r] * SCALE; if (kg0 > qg) v0 = NEGBIG;
      float v1 = s[1][r] * SCALE; if (kg1 > qg) v1 = NEGBIG;
      float mx = fmaxf(v0, v1);
#pragma unroll
      for (int msk = 1; msk <= 8; msk <<= 1)
        mx = fmaxf(mx, __shfl_xor(mx, msk, 32));
      float mNew  = fmaxf(mI[r], mx);
      float alpha = __expf(mI[r] - mNew);
      float e0 = __expf(v0 - mNew);
      float e1 = __expf(v1 - mNew);
      float rs = e0 + e1;
#pragma unroll
      for (int msk = 1; msk <= 8; msk <<= 1)
        rs += __shfl_xor(rs, msk, 32);
      lI[r] = lI[r] * alpha + rs;
      mI[r] = mNew;
#pragma unroll
      for (int dt = 0; dt < 4; ++dt) o[dt][r] *= alpha;
      p0[r] = e0; p1[r] = e1;
    }

    // transpose P (C-frag: lane=key) -> A-frag (lane=query) via per-wave LDS
#pragma unroll
    for (int r = 0; r < 8; ++r) {
      int row = r + 8 * g;
      myPs[row * 40 + ln]      = (_Float16)p0[r];
      myPs[row * 40 + 16 + ln] = (_Float16)p1[r];
    }
    h16 pa = load_frag_split(&myPs[ln * 40 + 8 * g],
                             &myPs[ln * 40 + 16 + 8 * g]);

    // P @ V: 4 d-column tiles of 16
#pragma unroll
    for (int dt = 0; dt < 4; ++dt) {
      h16 vb = load_frag(&Vt[(dt * 16 + ln) * 48 + 8 * g]);
      o[dt] = wmma_f16(pa, vb, o[dt]);
    }
    __syncthreads();
  }

  // normalize and write attn_vec (f32) : layout (b, h, t, d)
#pragma unroll
  for (int r = 0; r < 8; ++r) {
    float inv = 1.0f / lI[r];
    int q = q0 + r + 8 * g;
    size_t rowBase = ((size_t)bh * SEQ + q) * DH;
#pragma unroll
    for (int dt = 0; dt < 4; ++dt)
      attn[rowBase + dt * 16 + ln] = o[dt][r] * inv;
  }
}

// ---------------------------------------------------------------------------
// Kernel 3: head mean (bandwidth kernel), f16 result for the final WMMA GEMM.
// ---------------------------------------------------------------------------
__global__ __launch_bounds__(256) void head_mean_kernel(
    const float* __restrict__ attn, _Float16* __restrict__ Mh) {
  int idx = blockIdx.x * 256 + threadIdx.x;
  if (idx >= ROWS * DH) return;
  int row = idx >> 6, d = idx & 63;
  int b = row >> 11, t = row & (SEQ - 1);
  float s = 0.f;
#pragma unroll
  for (int h = 0; h < NH; ++h)
    s += attn[((size_t)((b * NH + h) * SEQ + t)) * DH + d];
  Mh[idx] = (_Float16)(s * 0.125f);
}

// ---------------------------------------------------------------------------
// Kernel 4: out(8192x512) = Mh(8192x64) @ Wout(64x512).  K=64 -> 2 WMMA steps.
// Wout^T staged once in LDS (f32->f16); A fragments loaded direct from global.
// ---------------------------------------------------------------------------
__global__ __launch_bounds__(256) void out_gemm_kernel(
    const _Float16* __restrict__ Mh, const float* __restrict__ Wout,
    float* __restrict__ out) {
  __shared__ _Float16 WoT[64 * 80];      // [n][k], pad 80

  const int tid  = threadIdx.x;
  const int wave = tid >> 5, lane = tid & 31;
  const int g = lane >> 4, ln = lane & 15;
  const int m0b = blockIdx.x * 128;
  const int n0b = blockIdx.y * 64;
  const int wm = wave >> 1, wn = wave & 1;

  for (int idx = tid; idx < 64 * 64; idx += 256) {
    int k = idx >> 6, n = idx & 63;
    WoT[n * 80 + k] = (_Float16)Wout[(size_t)k * HID + n0b + n];
  }
  __syncthreads();

  f8 acc[2][2] = {};
#pragma unroll
  for (int ks = 0; ks < 2; ++ks) {
    h16 a[2], b[2];
#pragma unroll
    for (int tm = 0; tm < 2; ++tm) {
      const _Float16* base =
          Mh + (size_t)(m0b + wm * 32 + tm * 16 + ln) * DH + ks * 32;
      a[tm] = load_frag_split(base + 8 * g, base + 16 + 8 * g);
    }
#pragma unroll
    for (int tn = 0; tn < 2; ++tn)
      b[tn] = load_frag(&WoT[(wn * 32 + tn * 16 + ln) * 80 + ks * 32 + 8 * g]);
#pragma unroll
    for (int tm = 0; tm < 2; ++tm)
#pragma unroll
      for (int tn = 0; tn < 2; ++tn)
        acc[tm][tn] = wmma_f16(a[tm], b[tn], acc[tm][tn]);
  }

#pragma unroll
  for (int tm = 0; tm < 2; ++tm)
#pragma unroll
    for (int tn = 0; tn < 2; ++tn)
#pragma unroll
      for (int r = 0; r < 8; ++r) {
        int grow = m0b + wm * 32 + tm * 16 + r + 8 * g;
        int gcol = n0b + wn * 32 + tn * 16 + ln;
        out[(size_t)grow * HID + gcol] = acc[tm][tn][r];
      }
}

// ---------------------------------------------------------------------------
extern "C" void kernel_launch(void* const* d_in, const int* in_sizes, int n_in,
                              void* d_out, int out_size, void* d_ws,
                              size_t ws_size, hipStream_t stream) {
  (void)in_sizes; (void)n_in; (void)out_size; (void)ws_size;
  const float* x    = (const float*)d_in[0];   // (4,2048,512)
  const float* Wqkv = (const float*)d_in[1];   // (512,1088)
  const float* Wout = (const float*)d_in[2];   // (64,512)

  float* out  = (float*)d_out;                       // (4,2048,512)
  float* attn = out + (size_t)ROWS * HID;            // (4,8,2048,64)

  _Float16* Qh = (_Float16*)d_ws;                    // (b,h,t,d) f16
  _Float16* Kh = Qh + (size_t)BSZ * NH * SEQ * DH;   // (b,h,t,d) f16
  _Float16* Vh = Kh + (size_t)BSZ * NH * SEQ * DH;   // (b,t,d)   f16
  _Float16* Mh = Vh + (size_t)BSZ * SEQ * DH;        // (b,t,d)   f16

  qkv_gemm_kernel<<<dim3(ROWS / 128, QKVN / 64), 256, 0, stream>>>(
      x, Wqkv, Qh, Kh, Vh);
  attn_kernel<<<dim3(BSZ * NH, SEQ / 128), 256, 0, stream>>>(Qh, Kh, Vh, attn);
  head_mean_kernel<<<(ROWS * DH + 255) / 256, 256, 0, stream>>>(attn, Mh);
  out_gemm_kernel<<<dim3(ROWS / 128, HID / 64), 256, 0, stream>>>(Mh, Wout, out);
}
